// LSTM_43765716746290
// MI455X (gfx1250) — matrix operand order repaired and checked by reading, not compiled, not run on hardware
//
#include <hip/hip_runtime.h>
#include <hip/hip_bf16.h>

// ---------------------------------------------------------------------------
// Persistent fused LSTM for MI455X (gfx1250, wave32, WMMA).
//   B=64, T=2048, D=128, F=256.  Weights live in LDS in bf16 for all 2048
//   steps; recurrent GEMM uses v_wmma_f32_16x16x32_bf16 (f32 accumulate);
//   h is exchanged between the 4 partitions through L2 each step.
//   Inner loop batches all LDS operand loads per K-tile ahead of the 4 gate
//   WMMAs so the scheduler can use partial s_wait_dscnt instead of draining
//   the DS pipe before every matrix op.
// ---------------------------------------------------------------------------

typedef __attribute__((ext_vector_type(16))) __bf16 v16bf;
typedef __attribute__((ext_vector_type(8)))  float  v8f;

#define Bsz 64
#define Tsz 2048
#define Dsz 128
#define Fsz 256
#define NWG 4
#define FS  64            // hidden units per workgroup
#define COLS 256          // 4 gates * FS gate-columns per workgroup
#define SH  (Fsz + 8)     // padded LDS row stride for h   (264 bf16)
#define SX  (Dsz + 8)     // padded LDS row stride for x_t (136 bf16)
#define SW  (Fsz + 8)     // padded row stride of Wh slice (K = 256)
#define SWI (Dsz + 8)     // padded row stride of Wi slice (K = 128)
#define THREADS 512       // 16 waves

// LDS layout (in bf16 elements)
#define OFF_WHS 0
#define OFF_WIS (COLS * SW)                 //  67584
#define OFF_H   (OFF_WIS + COLS * SWI)     // 102400
#define OFF_X   (OFF_H + Bsz * SH)         // 119296
#define SMEM_ELEMS (OFF_X + Bsz * SX)      // 128000 elems
#define SMEM_BYTES (SMEM_ELEMS * 2)        // 256000 B  (< 320 KB per WGP)

static __device__ __forceinline__ v8f wmma_bf16(v16bf a, v16bf b, v8f c) {
    return __builtin_amdgcn_wmma_f32_16x16x32_bf16(
        /*neg_a=*/false, a, /*neg_b=*/false, b,
        /*c_mod=*/(short)0, c, /*reuse_a=*/false, /*reuse_b=*/false);
}

// A operand: 16x32 bf16 tile from row-major LDS (row stride `stride` elems).
// Contiguous k order (same as ldB): lanes 0-15 -> rows 0-15 with K = kt*32 + 0..15;
//             lanes 16-31 -> same rows with K = kt*32 + 16..31.
static __device__ __forceinline__ v16bf ldA(const __bf16* base, int row0,
                                            int stride, int kt) {
    const int lane = threadIdx.x & 31;
    const int m    = lane & 15;
    const int k0   = kt * 32 + (lane >> 4) * 16;
    const __bf16* p = base + (row0 + m) * stride + k0;
    union { v16bf v; uint4 q[2]; } u;
    u.q[0] = *(const uint4*)(p);        // K = k0 .. k0+7
    u.q[1] = *(const uint4*)(p + 8);    // K = k0+8 .. k0+15
    return u.v;
}

// B operand: 32x16 bf16 tile from column-major LDS weight slab W[col][k].
// Contiguous k order (same as ldA): lane n (n<16) -> column n, K = kt*32 + 0..15;
//             lane 16+n     -> column n, K = kt*32 + 16..31. (contiguous 32B)
static __device__ __forceinline__ v16bf ldB(const __bf16* w, int col0,
                                            int stride, int kt) {
    const int lane = threadIdx.x & 31;
    const __bf16* p = w + (col0 + (lane & 15)) * stride
                        + kt * 32 + (lane >> 4) * 16;
    union { v16bf v; uint4 q[2]; } u;
    u.q[0] = *(const uint4*)(p);
    u.q[1] = *(const uint4*)(p + 8);
    return u.v;
}

static __device__ __forceinline__ float sigm(float x) {
    return 1.0f / (1.0f + __expf(-x));
}

// ---------------------------------------------------------------------------
__global__ void lstm_init_kernel(unsigned* gsync, unsigned* hbuf_u32,
                                 float* out, const float* bd) {
    const int tid = blockIdx.x * blockDim.x + threadIdx.x;
    if (tid == 0) *gsync = 0u;
    // zero both h double-buffers: 2 * 64*256 bf16 = 16384 u32
    if (tid < 16384) hbuf_u32[tid] = 0u;
    if (tid < Bsz) out[tid] = bd[0];
}

// ---------------------------------------------------------------------------
__global__ __launch_bounds__(THREADS, 1)
void lstm_persist_kernel(const float* __restrict__ x,
                         const float* __restrict__ Wi,
                         const float* __restrict__ Wh,
                         const float* __restrict__ bvec,
                         const float* __restrict__ Wd,
                         float* __restrict__ out,
                         __bf16* __restrict__ hbuf,   // 2 x [64][256] bf16
                         unsigned* __restrict__ gsync) {
    extern __shared__ __bf16 smem[];
    __bf16* whs = smem + OFF_WHS;   // [COLS][SW]  bf16, column-major weights
    __bf16* wis = smem + OFF_WIS;   // [COLS][SWI]
    __bf16* hls = smem + OFF_H;     // [64][SH]
    __bf16* xls = smem + OFF_X;     // [64][SX]

    const int tid  = threadIdx.x;
    const int lane = tid & 31;
    const int wave = tid >> 5;          // 0..15
    const int wg   = blockIdx.x;        // 0..3
    const int m0   = (wave & 3) * 16;   // batch-row tile
    const int n0   = (wave >> 2) * 16;  // hidden-col tile within FS

    // ---- one-time: stage bf16 weight slices into LDS, column-major --------
    // local col lc: gate g = lc>>6, unit j = lc&63 -> global col g*256+wg*64+j
    for (int idx = tid; idx < COLS * Dsz; idx += THREADS) {
        const int lc = idx / Dsz, k = idx % Dsz;
        const int gcol = (lc >> 6) * Fsz + wg * FS + (lc & 63);
        wis[lc * SWI + k] = (__bf16)Wi[(size_t)k * (4 * Fsz) + gcol];
    }
    for (int idx = tid; idx < COLS * Fsz; idx += THREADS) {
        const int lc = idx / Fsz, k = idx % Fsz;
        const int gcol = (lc >> 6) * Fsz + wg * FS + (lc & 63);
        whs[lc * SW + k] = (__bf16)Wh[(size_t)k * (4 * Fsz) + gcol];
    }
    __syncthreads();

    // per-lane constants: this lane's gate column & biases (loop-invariant)
    const int gc = wg * FS + n0 + (lane & 15);       // hidden unit 0..255
    const float bi = bvec[0 * Fsz + gc];
    const float bf_ = bvec[1 * Fsz + gc];
    const float bg = bvec[2 * Fsz + gc];
    const float bo = bvec[3 * Fsz + gc];

    float c[8];
#pragma unroll
    for (int r = 0; r < 8; ++r) c[r] = 0.0f;

    // staging indices (each thread moves a fixed 32-elem chunk of h,
    // and a fixed 16-float chunk of x_t)
    const int hidx = tid * 32;                 // 512*32 == 64*256
    const int xrow = tid >> 3;                 // 0..63
    const int xcol = (tid & 7) * 16;           // 0..112

    for (int t = 0; t < Tsz; ++t) {
        // ---- stage h_t (global bf16, double buffered) -> LDS -------------
        {
            const __bf16* hrd = hbuf + (size_t)(t & 1) * (Bsz * Fsz);
            const uint4* src = (const uint4*)(hrd + hidx);
            __bf16* dst = hls + (hidx >> 8) * SH + (hidx & 255);
            uint4 a0 = src[0], a1 = src[1], a2 = src[2], a3 = src[3];
            *(uint4*)(dst)      = a0;
            *(uint4*)(dst +  8) = a1;
            *(uint4*)(dst + 16) = a2;
            *(uint4*)(dst + 24) = a3;
        }
        // ---- stage x[:, t, :] f32 -> bf16 LDS ----------------------------
        {
            const float* xs = x + ((size_t)xrow * Tsz + t) * Dsz + xcol;
            union { uint4 q; __bf16 e[8]; } u0, u1;
#pragma unroll
            for (int j = 0; j < 8; ++j) u0.e[j] = (__bf16)xs[j];
#pragma unroll
            for (int j = 0; j < 8; ++j) u1.e[j] = (__bf16)xs[8 + j];
            __bf16* dx = xls + xrow * SX + xcol;
            *(uint4*)(dx)     = u0.q;
            *(uint4*)(dx + 8) = u1.q;
            if (t + 1 < Tsz)   // prefetch next timestep (global_prefetch_b8)
                __builtin_prefetch(xs + Dsz, 0, 1);
        }
        __syncthreads();

        // ---- z = [x_t h_t] @ [Wi; Wh] for 4 gates (bf16 WMMA, f32 acc) ---
        // All operand loads for a K-tile are issued before its 4 WMMAs so
        // the DS latency overlaps XDL execution (partial s_wait_dscnt).
        v8f zi = {0,0,0,0,0,0,0,0}, zf = {0,0,0,0,0,0,0,0};
        v8f zg = {0,0,0,0,0,0,0,0}, zo = {0,0,0,0,0,0,0,0};
#pragma unroll
        for (int kt = 0; kt < Dsz / 32; ++kt) {          // x part, K=128
            const v16bf a  = ldA(xls, m0, SX, kt);
            const v16bf b0 = ldB(wis,   0 + n0, SWI, kt);
            const v16bf b1 = ldB(wis,  64 + n0, SWI, kt);
            const v16bf b2 = ldB(wis, 128 + n0, SWI, kt);
            const v16bf b3 = ldB(wis, 192 + n0, SWI, kt);
            zi = wmma_bf16(a, b0, zi);
            zf = wmma_bf16(a, b1, zf);
            zg = wmma_bf16(a, b2, zg);
            zo = wmma_bf16(a, b3, zo);
        }
#pragma unroll
        for (int kt = 0; kt < Fsz / 32; ++kt) {          // h part, K=256
            const v16bf a  = ldA(hls, m0, SH, kt);
            const v16bf b0 = ldB(whs,   0 + n0, SW, kt);
            const v16bf b1 = ldB(whs,  64 + n0, SW, kt);
            const v16bf b2 = ldB(whs, 128 + n0, SW, kt);
            const v16bf b3 = ldB(whs, 192 + n0, SW, kt);
            zi = wmma_bf16(a, b0, zi);
            zf = wmma_bf16(a, b1, zf);
            zg = wmma_bf16(a, b2, zg);
            zo = wmma_bf16(a, b3, zo);
        }

        // ---- gates + cell update; write new h (bf16) to other buffer -----
        // C/D layout: VGPR r, lanes 0-15 -> (M=r, N=lane); 16-31 -> (M=8+r).
        __bf16* hwr = hbuf + (size_t)((t + 1) & 1) * (Bsz * Fsz);
        const int rowbase = m0 + ((lane >> 4) * 8);
        float hnv[8];
#pragma unroll
        for (int r = 0; r < 8; ++r) {
            float iv = sigm(zi[r] + bi);
            float fv = sigm(zf[r] + bf_);
            float gv = tanhf(zg[r] + bg);
            float ov = sigm(zo[r] + bo);
            float cn = fv * c[r] + iv * gv;
            c[r] = cn;
            float hn = ov * tanhf(cn);
            hnv[r] = hn;
            hwr[(size_t)(rowbase + r) * Fsz + gc] = (__bf16)hn;
        }

        // ---- last step: fused Dense(1) from f32 h tiles ------------------
        if (t == Tsz - 1) {
            const float wd = Wd[gc];
#pragma unroll
            for (int r = 0; r < 8; ++r) {
                float p = hnv[r] * wd;
                p += __shfl_xor(p, 1, 16);
                p += __shfl_xor(p, 2, 16);
                p += __shfl_xor(p, 4, 16);
                p += __shfl_xor(p, 8, 16);
                if ((lane & 15) == 0) atomicAdd(&out[rowbase + r], p);
            }
        }

        // ---- grid-wide step barrier (4 participants) ---------------------
        __builtin_amdgcn_fence(__ATOMIC_RELEASE, "agent");
        __syncthreads();
        if (tid == 0) {
            __hip_atomic_fetch_add(gsync, 1u, __ATOMIC_RELAXED,
                                   __HIP_MEMORY_SCOPE_AGENT);
            const unsigned target = (unsigned)(NWG * (t + 1));
            while (__hip_atomic_load(gsync, __ATOMIC_RELAXED,
                                     __HIP_MEMORY_SCOPE_AGENT) < target) {
                __builtin_amdgcn_s_sleep(1);
            }
        }
        __syncthreads();
        __builtin_amdgcn_fence(__ATOMIC_ACQUIRE, "agent");
    }
}

// ---------------------------------------------------------------------------
extern "C" void kernel_launch(void* const* d_in, const int* in_sizes, int n_in,
                              void* d_out, int out_size, void* d_ws,
                              size_t ws_size, hipStream_t stream) {
    (void)in_sizes; (void)n_in; (void)out_size; (void)ws_size;
    const float* x    = (const float*)d_in[0];
    const float* Wi   = (const float*)d_in[1];
    const float* Wh   = (const float*)d_in[2];
    const float* bvec = (const float*)d_in[3];
    const float* Wd   = (const float*)d_in[4];
    const float* bd   = (const float*)d_in[5];
    float* out = (float*)d_out;

    // workspace: [0..3] barrier counter; [256 ..] 2 x [64][256] bf16 h buffers
    unsigned* gsync = (unsigned*)d_ws;
    __bf16*   hbuf  = (__bf16*)((char*)d_ws + 256);

    lstm_init_kernel<<<dim3(64), dim3(256), 0, stream>>>(
        gsync, (unsigned*)hbuf, out, bd);

    lstm_persist_kernel<<<dim3(NWG), dim3(THREADS), SMEM_BYTES, stream>>>(
        x, Wi, Wh, bvec, Wd, out, hbuf, gsync);
}
